// LSTMPredictor_40553081209541
// MI455X (gfx1250) — matrix operand order repaired and checked by reading, not compiled, not run on hardware
//
#include <hip/hip_runtime.h>
#include <hip/hip_bf16.h>
#include <math.h>

// Problem constants (from reference)
#define BATCH 1024
#define TLEN  512
#define IDIM  4
#define HDIM  128
#define GDIM  (4 * HDIM)   // 512 gate pre-activations
#define ODIM  4

// Tiling
#define ROWS_PER_WG 16         // batch rows per workgroup
#define WAVES_PER_WG 8         // 8 waves * 4 gates * 16 cols = 512 gate columns
#define THREADS (WAVES_PER_WG * 32)
#define NUM_WG (BATCH / ROWS_PER_WG)   // 64 workgroups
#define HPAD 136               // halves per LDS h-row (272B, 16B aligned, bank-spread)

typedef __attribute__((ext_vector_type(16))) _Float16 v16h;
typedef __attribute__((ext_vector_type(8)))  _Float16 v8h;
typedef __attribute__((ext_vector_type(8)))  float    v8f;

__device__ __forceinline__ float sigmoidf_fast(float x) {
    return 1.0f / (1.0f + __expf(-x));
}
__device__ __forceinline__ float tanhf_fast(float x) {
    // tanh(x) = 1 - 2/(exp(2x)+1)
    return 1.0f - 2.0f / (__expf(2.0f * x) + 1.0f);
}

__global__ __launch_bounds__(THREADS)
void lstm_persistent_kernel(const float* __restrict__ x,
                            const float* __restrict__ W_ih,
                            const float* __restrict__ W_hh,
                            const float* __restrict__ b_ih,
                            const float* __restrict__ b_hh,
                            const float* __restrict__ W_fc,
                            const float* __restrict__ b_fc,
                            float* __restrict__ out) {
    // Double-buffered h (f16, WMMA-A friendly layout): one barrier per timestep.
    __shared__ __align__(16) _Float16 h_sh[2][ROWS_PER_WG * HPAD];
    __shared__ __align__(16) float    x_sh[2][ROWS_PER_WG * IDIM]; // x prefetch buffers

    const int tid  = threadIdx.x;
    const int lane = tid & 31;
    const int wave = tid >> 5;
    const int r0   = blockIdx.x * ROWS_PER_WG;

    const int nlo  = lane & 15;          // N within tile / M for A-fragment
    const int hi16 = lane >> 4;          // 0 for lanes 0-15, 1 for lanes 16-31
    const int hhw  = wave * 16 + nlo;    // this lane's hidden-unit column [0,128)
    const int mA   = nlo;                // A-fragment row (batch row in tile)
    const int klo  = hi16 * 16;          // A-fragment K sub-offset (halves)

    // ---- Preload per-lane constants: W_ih rows + combined bias for the 4 gates ----
    float4 wih[4];
    float  bias2[4];
#pragma unroll
    for (int q = 0; q < 4; ++q) {
        const int n = q * HDIM + hhw;                  // gate column in [0, 512)
        wih[q]   = *reinterpret_cast<const float4*>(W_ih + n * IDIM);
        bias2[q] = b_ih[n] + b_hh[n];
    }

    // ---- Preload all W_hh B-fragments (f16) into registers: 4 gates x 4 K-chunks ----
    // B k order chosen here (contiguous, matches the A load): VGPR v, lanes 0-15: (K=2v,2v+1, N=lane); lanes 16-31: K+=16
    v16h Bf[4][4];
    const int kgrp = hi16 * 16;
#pragma unroll
    for (int q = 0; q < 4; ++q) {
        const int n = q * HDIM + hhw;
#pragma unroll
        for (int kk = 0; kk < 4; ++kk) {
            union { v16h v; _Float16 e[16]; } u;
#pragma unroll
            for (int v = 0; v < 8; ++v) {
                const int k0 = kk * 32 + kgrp + 2 * v;
                u.e[2 * v]     = (_Float16)W_hh[n * HDIM + k0];
                u.e[2 * v + 1] = (_Float16)W_hh[n * HDIM + k0 + 1];
            }
            Bf[q][kk] = u.v;
        }
    }

    // ---- Zero initial state; prefetch x[t=0] ----
    for (int idx = tid; idx < ROWS_PER_WG * HPAD; idx += THREADS) h_sh[0][idx] = (_Float16)0.0f;
    if (tid < ROWS_PER_WG * IDIM) {
        const int m = tid >> 2, i = tid & 3;
        x_sh[0][tid] = x[(size_t)(r0 + m) * TLEN * IDIM + i];
    }
    float c_reg[8];
#pragma unroll
    for (int v = 0; v < 8; ++v) c_reg[v] = 0.0f;

    // ---- Time loop: 512 sequential steps, one barrier per step, no global sync ----
    for (int t = 0; t < TLEN; ++t) {
        const int rd = t & 1;        // h / x buffer to consume this step
        const int wr = rd ^ 1;       // h buffer to produce / x buffer to prefetch

        __syncthreads();  // h[rd] writes (step t-1) and x[rd] staging now visible

        // Prefetch x[b, t+1, :] into the opposite buffer (hidden behind GEMM)
        if (t + 1 < TLEN && tid < ROWS_PER_WG * IDIM) {
            const int m = tid >> 2, i = tid & 3;
            x_sh[wr][tid] = x[((size_t)(r0 + m) * TLEN + (t + 1)) * IDIM + i];
        }

        // gates_pre = h_t @ W_hh^T  (per wave: 4 N-tiles covering hh columns
        //             [16w,16w+16) for gates i,f,g,o), f32 accumulate
        v8f C0 = {}, C1 = {}, C2 = {}, C3 = {};
#pragma unroll
        for (int kk = 0; kk < 4; ++kk) {
            union { v16h v; v8h h2[2]; } A;
            const int base = mA * HPAD + kk * 32 + klo;
            A.h2[0] = *reinterpret_cast<const v8h*>(&h_sh[rd][base]);
            A.h2[1] = *reinterpret_cast<const v8h*>(&h_sh[rd][base + 8]);
            C0 = __builtin_amdgcn_wmma_f32_16x16x32_f16(false, A.v, false, Bf[0][kk], (short)0, C0, false, false);
            C1 = __builtin_amdgcn_wmma_f32_16x16x32_f16(false, A.v, false, Bf[1][kk], (short)0, C1, false, false);
            C2 = __builtin_amdgcn_wmma_f32_16x16x32_f16(false, A.v, false, Bf[2][kk], (short)0, C2, false, false);
            C3 = __builtin_amdgcn_wmma_f32_16x16x32_f16(false, A.v, false, Bf[3][kk], (short)0, C3, false, false);
        }

        // Elementwise LSTM cell update; each slot (v,lane) = (row m, hidden hh).
        // Writes go to h[wr] — no second barrier needed (readers used h[rd]).
#pragma unroll
        for (int v = 0; v < 8; ++v) {
            const int m = v + (hi16 << 3);
            const float4 xm = *reinterpret_cast<const float4*>(&x_sh[rd][m * 4]);
            const float gi = C0[v] + bias2[0] + xm.x * wih[0].x + xm.y * wih[0].y + xm.z * wih[0].z + xm.w * wih[0].w;
            const float gf = C1[v] + bias2[1] + xm.x * wih[1].x + xm.y * wih[1].y + xm.z * wih[1].z + xm.w * wih[1].w;
            const float gg = C2[v] + bias2[2] + xm.x * wih[2].x + xm.y * wih[2].y + xm.z * wih[2].z + xm.w * wih[2].w;
            const float go = C3[v] + bias2[3] + xm.x * wih[3].x + xm.y * wih[3].y + xm.z * wih[3].z + xm.w * wih[3].w;
            const float ig = sigmoidf_fast(gi);
            const float fg = sigmoidf_fast(gf);
            const float g2 = tanhf_fast(gg);
            const float og = sigmoidf_fast(go);
            c_reg[v] = fg * c_reg[v] + ig * g2;
            const float hn = og * tanhf_fast(c_reg[v]);
            h_sh[wr][m * HPAD + hhw] = (_Float16)hn;
        }
    }
    __syncthreads();

    // ---- FC head: out[b, o] = h_T[b] . W_fc[o] + b_fc[o] (64 values per WG).
    //      Final h landed in buffer ((TLEN - 1) & 1) ^ 1 == 0.
    if (tid < ROWS_PER_WG * ODIM) {
        const int m = tid >> 2, o = tid & 3;
        float acc = b_fc[o];
#pragma unroll 4
        for (int k = 0; k < HDIM; ++k)
            acc += (float)h_sh[0][m * HPAD + k] * W_fc[o * HDIM + k];
        out[(size_t)(r0 + m) * ODIM + o] = acc;
    }
}

extern "C" void kernel_launch(void* const* d_in, const int* in_sizes, int n_in,
                              void* d_out, int out_size, void* d_ws, size_t ws_size,
                              hipStream_t stream) {
    (void)in_sizes; (void)n_in; (void)d_ws; (void)ws_size; (void)out_size;
    const float* x    = (const float*)d_in[0];
    const float* W_ih = (const float*)d_in[1];
    const float* W_hh = (const float*)d_in[2];
    const float* b_ih = (const float*)d_in[3];
    const float* b_hh = (const float*)d_in[4];
    const float* W_fc = (const float*)d_in[5];
    const float* b_fc = (const float*)d_in[6];
    float* out = (float*)d_out;

    lstm_persistent_kernel<<<NUM_WG, THREADS, 0, stream>>>(
        x, W_ih, W_hh, b_ih, b_hh, W_fc, b_fc, out);
}
